// LSTMCellWrapper_266287972945
// MI455X (gfx1250) — matrix operand order repaired and checked by reading, not compiled, not run on hardware
//
#include <hip/hip_runtime.h>

// LSTM forward, MI455X (gfx1250, wave32, WMMA f32<=f16).
// SEQ=512, B=64, IN=H=1024.
//
// Per step: gates(64x4096) = [x_t | h_{t-1}] (64x2048) @ Wcat^T (2048x4096) + b
// Grid 64x4 blocks of 256 threads (8 waves). Wave w = (gate w>>1, K-half w&1):
// 32 WMMAs split over two accumulator chains of 16, K-loop unrolled 4x so the
// scheduler can rotate A/B fragment registers (avoids WMMA->VALU WAR NOPs).
// Gate tiles + K-half partials meet in LDS; pointwise cell update finishes the
// step in-block, so each timestep is one kernel (512 graph nodes, no grid sync).

#define SEQ   512
#define BATCH 64
#define INP   1024
#define HID   1024
#define GATES 4096   // 4*HID
#define KTOT  2048   // INP + HID

typedef _Float16 v16h __attribute__((ext_vector_type(16)));
typedef _Float16 v8h  __attribute__((ext_vector_type(8)));
typedef float    v8f  __attribute__((ext_vector_type(8)));

__device__ __forceinline__ float sigmoidf_(float x) {
    return 1.0f / (1.0f + __expf(-x));
}

// 16x32 f16 A-fragment from a row-major f32 row (contiguous k order, same as load_b_f16):
// lane l (g = l>>4): halves 0..7  <- row[16g + 0..7]
//                    halves 8..15 <- row[16g + 8 + 0..7]
__device__ __forceinline__ v16h load_a_f32(const float* __restrict__ row, int g) {
    const float4* p0 = (const float4*)(row + 16 * g);
    const float4* p1 = (const float4*)(row + 16 * g + 8);
    float4 a0 = p0[0], a1 = p0[1];
    float4 b0 = p1[0], b1 = p1[1];
    v16h a;
    a[0]  = (_Float16)a0.x; a[1]  = (_Float16)a0.y; a[2]  = (_Float16)a0.z; a[3]  = (_Float16)a0.w;
    a[4]  = (_Float16)a1.x; a[5]  = (_Float16)a1.y; a[6]  = (_Float16)a1.z; a[7]  = (_Float16)a1.w;
    a[8]  = (_Float16)b0.x; a[9]  = (_Float16)b0.y; a[10] = (_Float16)b0.z; a[11] = (_Float16)b0.w;
    a[12] = (_Float16)b1.x; a[13] = (_Float16)b1.y; a[14] = (_Float16)b1.z; a[15] = (_Float16)b1.w;
    return a;
}

// 32x16 f16 B-fragment: lane l holds column n = l&15, 16 contiguous halfs
// K = K0 + 16*(l>>4) + 0..15 (two 16-byte loads).
__device__ __forceinline__ v16h load_b_f16(const _Float16* __restrict__ wrow) {
    const v8h* q = (const v8h*)wrow;
    v8h lo = q[0], hi = q[1];
    v16h b;
#pragma unroll
    for (int p = 0; p < 8; ++p) { b[p] = lo[p]; b[8 + p] = hi[p]; }
    return b;
}

__global__ __launch_bounds__(256) void lstm_step(
    const float* __restrict__ x,        // (B,S,I) flat
    const float* __restrict__ hprev,    // batch row b at hprev + b*hstride
    long long hstride,
    const float* __restrict__ cin,      // (B,H)
    float* __restrict__ cout,           // (B,H)
    float* __restrict__ outbase,        // main output, physically (B,S,H)
    float* __restrict__ hN,             // final-h slot (only at t==SEQ-1)
    const _Float16* __restrict__ Wcat,  // (4096, 2048) f16: [W_ih | W_hh]
    const float* __restrict__ bias,     // (4096) = b_ih + b_hh
    int t)
{
    const int w    = threadIdx.x >> 5;   // wave 0..7
    const int gate = w >> 1;             // 0..3 -> i,f,g,o
    const int kh   = w & 1;              // 0: x half of K, 1: h half of K
    const int l    = threadIdx.x & 31;   // lane
    const int g    = l >> 4;             // lane half-group
    const int jt   = blockIdx.x;         // 0..63  (H column tile)
    const int mt   = blockIdx.y;         // 0..3   (batch tile)

    const int b      = mt * 16 + (l & 15);     // A row (batch) this lane loads
    const int nlocal = jt * 16 + (l & 15);     // column within H
    const int ncol   = gate * HID + nlocal;    // column within 4H

    // A source: K-half 0 reads x_t, K-half 1 reads h_{t-1}.
    const float* arow = kh ? (hprev + (size_t)b * hstride)
                           : (x + (size_t)b * SEQ * INP + (size_t)t * INP);
    const _Float16* wrow = Wcat + (size_t)ncol * KTOT + (size_t)kh * INP + 16 * g;

    // Two accumulator chains (depth 16 each) over this wave's K range of 1024.
    // Unroll 4x (8 WMMAs / body) so fragment registers rotate across
    // iterations: no WMMA->VALU WAR hazard NOPs, deeper load pipelining.
    v8f acc0 = {}, acc1 = {};
#pragma unroll 4
    for (int k0 = 0; k0 < 1024; k0 += 64) {
        v16h a0 = load_a_f32(arow + k0, g);
        v16h b0 = load_b_f16(wrow + k0);
        acc0 = __builtin_amdgcn_wmma_f32_16x16x32_f16(false, a0, false, b0,
                                                      (short)0, acc0, false, false);
        v16h a1 = load_a_f32(arow + k0 + 32, g);
        v16h b1 = load_b_f16(wrow + k0 + 32);
        acc1 = __builtin_amdgcn_wmma_f32_16x16x32_f16(false, a1, false, b1,
                                                      (short)0, acc1, false, false);
    }

    const float bv = kh ? 0.0f : bias[ncol];   // add bias exactly once per gate

    // Stage 8 partial 16x16 tiles in LDS. C/D layout: VGPR r = (M = r+8g, N = l&15).
    __shared__ float tiles[8][256];
#pragma unroll
    for (int r = 0; r < 8; ++r) {
        int M = r + 8 * g;
        tiles[w][M * 16 + (l & 15)] = acc0[r] + acc1[r] + bv;
    }
    __syncthreads();

    // Pointwise LSTM cell update: 256 elements, one per thread.
    {
        int e  = threadIdx.x;
        int ml = e >> 4;                  // local batch row
        int jl = e & 15;                  // local column
        int bb = mt * 16 + ml;
        int j  = jt * 16 + jl;
        float gi = tiles[0][e] + tiles[1][e];
        float gf = tiles[2][e] + tiles[3][e];
        float gg = tiles[4][e] + tiles[5][e];
        float go = tiles[6][e] + tiles[7][e];
        float cold = cin[(size_t)bb * HID + j];
        float cnew = sigmoidf_(gf) * cold + sigmoidf_(gi) * tanhf(gg);
        float hh   = sigmoidf_(go) * tanhf(cnew);
        cout[(size_t)bb * HID + j] = cnew;
        outbase[(size_t)bb * SEQ * HID + (size_t)t * HID + j] = hh;
        if (hN) hN[(size_t)bb * HID + j] = hh;
    }
}

// Convert + concatenate weights: Wcat[n][k] = f16( k<1024 ? W_ih[n][k] : W_hh[n][k-1024] )
__global__ __launch_bounds__(256) void prep_wcat(const float* __restrict__ Wih,
                                                 const float* __restrict__ Whh,
                                                 _Float16* __restrict__ Wcat)
{
    int idx = blockIdx.x * 256 + threadIdx.x;   // 4096*2048 = 8388608 total
    int n = idx >> 11;
    int k = idx & 2047;
    float v = (k < INP) ? Wih[(size_t)n * INP + k]
                        : Whh[(size_t)n * HID + (k - INP)];
    Wcat[idx] = (_Float16)v;
}

__global__ __launch_bounds__(256) void prep_bias(const float* __restrict__ bih,
                                                 const float* __restrict__ bhh,
                                                 float* __restrict__ bias)
{
    int i = blockIdx.x * 256 + threadIdx.x;
    if (i < GATES) bias[i] = bih[i] + bhh[i];
}

extern "C" void kernel_launch(void* const* d_in, const int* in_sizes, int n_in,
                              void* d_out, int out_size, void* d_ws, size_t ws_size,
                              hipStream_t stream) {
    (void)in_sizes; (void)n_in; (void)out_size; (void)ws_size;

    const float* inp = (const float*)d_in[0];   // (B,S,I)
    const float* h0  = (const float*)d_in[1];   // (1,B,H)
    const float* c0  = (const float*)d_in[2];   // (1,B,H)
    const float* Wih = (const float*)d_in[3];   // (4H,I)
    const float* Whh = (const float*)d_in[4];   // (4H,H)
    const float* bih = (const float*)d_in[5];   // (4H)
    const float* bhh = (const float*)d_in[6];   // (4H)

    float* out   = (float*)d_out;                         // (B,S,H) physically
    float* hN    = out + (size_t)SEQ * BATCH * HID;       // final h (B*H)
    float* cslot = hN + (size_t)BATCH * HID;              // final c (B*H), live c state

    char*     ws   = (char*)d_ws;
    _Float16* Wcat = (_Float16*)ws;                                  // 16 MB
    float*    bias = (float*)(ws + (size_t)GATES * KTOT * sizeof(_Float16));

    prep_wcat<<<(GATES * KTOT) / 256, 256, 0, stream>>>(Wih, Whh, Wcat);
    prep_bias<<<GATES / 256, 256, 0, stream>>>(bih, bhh, bias);

    dim3 grid(HID / 16, BATCH / 16);   // 64 x 4
    dim3 block(256);                   // 8 waves of 32
    for (int t = 0; t < SEQ; ++t) {
        // h_{t-1} read back from its final location in d_out; batch stride S*H.
        const float* hprev   = (t == 0) ? h0 : (out + (size_t)(t - 1) * HID);
        long long    hstride = (t == 0) ? (long long)HID : (long long)SEQ * HID;
        const float* cin     = (t == 0) ? c0 : cslot;
        float*       hNarg   = (t == SEQ - 1) ? hN : nullptr;
        lstm_step<<<grid, block, 0, stream>>>(inp, hprev, hstride, cin, cslot,
                                              out, hNarg, Wcat, bias, t);
    }
}
